// PegasusXGlobalLocalAttention_27934467293315
// MI455X (gfx1250) — matrix operand order repaired and checked by reading, not compiled, not run on hardware
//
#include <hip/hip_runtime.h>

// ---- problem constants (PegasusX global/local attention) ----
#define BB 2
#define SS 4096
#define GG 128
#define DD 1024
#define HH 16
#define FF 64          // head dim
#define KKB 512        // BLOCK_SIZE
#define SCALING 0.125f // 64^-0.5

typedef __attribute__((ext_vector_type(16))) __bf16 v16bf;
typedef __attribute__((ext_vector_type(8)))  float  v8f;
typedef __attribute__((ext_vector_type(4)))  unsigned int v4u;
typedef __attribute__((ext_vector_type(8)))  int v8i_t;
typedef __attribute__((ext_vector_type(4)))  int v4i_t;

// float -> bf16 (round-to-nearest-even), returned as raw u16
__device__ __forceinline__ unsigned short f2bf(float x) {
    union { float f; unsigned u; } v; v.f = x;
    unsigned r = (v.u + 0x7FFFu + ((v.u >> 16) & 1u)) >> 16;
    return (unsigned short)r;
}

// A-fragment (16x32 bf16): lane holds row = lane&15; its 16 values are
// K = [16h .. 16h+15]  (h = lane>>4)  -> two 16B loads.
__device__ __forceinline__ v16bf load_afrag(const __bf16* base, int half) {
    v16bf r;
    uint4 a, b;
    __builtin_memcpy(&a, base + 16 * half, 16);
    __builtin_memcpy(&b, base + 16 * half + 8, 16);
    __builtin_memcpy(&r, &a, 16);
    __builtin_memcpy((char*)&r + 16, &b, 16);
    return r;
}

// B-fragment (32x16 bf16): lane holds col = lane&15; K = 16h..16h+15
// contiguous -> one 32B load (caller passes pointer already offset by 16h).
__device__ __forceinline__ v16bf load_bfrag(const __bf16* p) {
    v16bf r;
    uint4 a, b;
    const uint4* p4 = (const uint4*)p;
    a = p4[0]; b = p4[1];
    __builtin_memcpy(&r, &a, 16);
    __builtin_memcpy((char*)&r + 16, &b, 16);
    return r;
}

__device__ __forceinline__ v8f wmma_bf16(v16bf a, v16bf b, v8f c) {
    return __builtin_amdgcn_wmma_f32_16x16x32_bf16(false, a, false, b, (short)0, c,
                                                   false, false);
}

// ---------------- f32 -> bf16 conversion ----------------
__global__ void k_cvt_bf16(const float* __restrict__ src,
                           unsigned short* __restrict__ dst, int n) {
    int i = blockIdx.x * blockDim.x + threadIdx.x;
    if (i < n) dst[i] = f2bf(src[i]);
}

// ---------------- WMMA GEMM: C = A[M,Kd] * W[N,Kd]^T ----------------
// 256 threads = 8 waves. The block's 16-row A tile (16 x Kd bf16, 32KB) is
// staged into LDS with ONE Tensor-Data-Mover descriptor issued by wave 0
// (tensor_load_to_lds + s_wait_tensorcnt), then shared by all 8 waves.
// Each wave computes a 16x64 strip of C: 4 accumulators reusing one
// A-fragment per 32-K step (4 WMMAs per A load).
// mode 0: bf16 out, scatter [B,H,L,F]      (Q: scale=0.125, K: scale=1)
// mode 1: bf16 out, scatter [B,H,F,L]      (V transposed for P*V B-frags)
// mode 2: f32 out, row-major [M,Nn]        (final Wo projection -> d_out)
__global__ void __launch_bounds__(256)
k_gemm_bf16_wmma(const __bf16* __restrict__ A,
                 const __bf16* __restrict__ W,
                 void* __restrict__ outp,
                 int Kd, int Nn, int Lsh, int mode, float scale) {
    __shared__ __align__(128) unsigned short lds_a[16 * DD]; // 16 rows x Kd bf16

    const int lane = threadIdx.x & 31;
    const int wv   = threadIdx.x >> 5;
    const int half = lane >> 4;
    const int l15  = lane & 15;
    const int m0   = blockIdx.y * 16;
    const int n0   = (blockIdx.x * 32 + wv * 4) * 16; // wave's first of 4 n-tiles

    // ---- TDM: stage A[m0..m0+15, 0..Kd) into LDS (one descriptor) ----
    if (wv == 0) {
        unsigned long long gaddr =
            (unsigned long long)(uintptr_t)(A + (size_t)m0 * Kd);
        // D# group 0: count=1, lds_addr=0 (lds_a is the only LDS object),
        // global_addr[56:0], type=2 at bits [127:126]
        v4u g0 = { 1u,
                   0u,
                   (unsigned)(gaddr & 0xFFFFFFFFu),
                   (unsigned)((gaddr >> 32) & 0x01FFFFFFu) | (2u << 30) };
        // D# group 1: wg_mask=0, data_size=1 (2B), tensor 2D: dim0=Kd,
        // dim1=16, tile0=Kd, tile1=16, dim0_stride=Kd
        const unsigned td0 = (unsigned)Kd, td1 = 16u;
        const unsigned tl0 = (unsigned)Kd, tl1 = 16u;
        const unsigned long long st0 = (unsigned long long)Kd;
        v8i_t g1 = { (int)(1u << 16),
                     (int)((td0 & 0xFFFFu) << 16),
                     (int)(((td0 >> 16) & 0xFFFFu) | ((td1 & 0xFFFFu) << 16)),
                     (int)(((td1 >> 16) & 0xFFFFu) | ((tl0 & 0xFFFFu) << 16)),
                     (int)(tl1 & 0xFFFFu),
                     (int)(st0 & 0xFFFFFFFFull),
                     (int)((st0 >> 32) & 0xFFFFull),
                     0 };
        v4i_t gz = { 0, 0, 0, 0 };
#if __clang_major__ >= 23
        v8i_t gz8 = { 0, 0, 0, 0, 0, 0, 0, 0 };
        __builtin_amdgcn_tensor_load_to_lds(g0, g1, gz, gz, gz8, 0);
#else
        __builtin_amdgcn_tensor_load_to_lds(g0, g1, gz, gz, 0);
#endif
        __builtin_amdgcn_s_wait_tensorcnt(0);
    }
    __syncthreads();
    // The TDM descriptor carries the LDS destination as an integer, so the
    // optimizer cannot see that lds_a was written. Escape the pointer and
    // clobber memory so the A-fragment loads below really read LDS.
    {
        unsigned short* lp = lds_a;
        asm volatile("" : "+v"(lp) :: "memory");
    }

    const __bf16* la  = (const __bf16*)lds_a + (size_t)l15 * Kd; // A row in LDS
    const __bf16* wp  = W + (size_t)(n0 + l15) * Kd + half * 16; // B frag base

    v8f acc[4] = {};
    for (int k0 = 0; k0 < Kd; k0 += 32) {
        __builtin_prefetch(wp + k0 + 256, 0, 1); // global_prefetch_b8
        v16bf a  = load_afrag(la + k0, half);
        v16bf b0 = load_bfrag(wp + k0);
        v16bf b1 = load_bfrag(wp + 16 * Kd + k0);
        v16bf b2 = load_bfrag(wp + 32 * Kd + k0);
        v16bf b3 = load_bfrag(wp + 48 * Kd + k0);
        acc[0] = wmma_bf16(a, b0, acc[0]);
        acc[1] = wmma_bf16(a, b1, acc[1]);
        acc[2] = wmma_bf16(a, b2, acc[2]);
        acc[3] = wmma_bf16(a, b3, acc[3]);
    }

#pragma unroll
    for (int t = 0; t < 4; ++t) {
        const int ncol = n0 + t * 16 + l15;
        if (mode == 2) {
            float* out = (float*)outp;
#pragma unroll
            for (int j = 0; j < 8; ++j) {
                int r = m0 + j + 8 * half;
                out[(size_t)r * Nn + ncol] = acc[t][j] * scale;
            }
        } else {
            unsigned short* out = (unsigned short*)outp;
            const int h = ncol >> 6;   // head
            const int f = ncol & 63;   // feature within head
            const int Lm = (1 << Lsh) - 1;
#pragma unroll
            for (int j = 0; j < 8; ++j) {
                int r = m0 + j + 8 * half;
                int b = r >> Lsh, l = r & Lm;  // L is a power of two
                size_t dst = (mode == 0)
                                 ? ((((size_t)b * HH + h) << Lsh | l) * FF + f)
                                 : ((((size_t)b * HH + h) * FF + f) << Lsh | l);
                out[dst] = f2bf(acc[t][j] * scale);
            }
        }
    }
}

// ---------------- flash-style attention over [global ; local] keys ----------
// One wave (32 threads) per 16-query tile. Keys streamed in chunks of 32:
// two 16-key score tiles via WMMA, online softmax (shfl_xor row reductions in
// C-layout), P transposed through LDS into an A-fragment, P*V via WMMA against
// transposed V. Output context written bf16 to [B, Lq, D].
__global__ void k_attn_wmma(const __bf16* __restrict__ Q,    // [B,H,Lq,F] (scaled)
                            const __bf16* __restrict__ Kg,   // [B,H,G,F]
                            const __bf16* __restrict__ Vgt,  // [B,H,F,G]
                            const __bf16* __restrict__ Kl,   // [B,H,S,F]
                            const __bf16* __restrict__ Vlt,  // [B,H,F,S]
                            const float*  __restrict__ mask, // [B,S]
                            unsigned short* __restrict__ ctx,// bf16 [B,Lq,D]
                            int Lq, int localLen, int blocked) {
    __shared__ unsigned short lds_p[16 * 32]; // P tile, bf16 bits

    const int lane = threadIdx.x & 31;
    const int half = lane >> 4;
    const int l15  = lane & 15;

    const int qtiles = Lq / 16;
    const int id = blockIdx.x;
    const int qt = id % qtiles;
    const int bh = id / qtiles;          // b*H + h
    const int h  = bh % HH;
    const int b  = bh / HH;
    const int q0 = qt * 16;
    const int lbase = blocked ? (q0 / KKB) * KKB : 0; // local-key base in S

    const __bf16* qrow = Q + ((size_t)bh * Lq + q0 + l15) * FF;
    const v16bf qa0 = load_afrag(qrow, half);       // f = 0..31
    const v16bf qa1 = load_afrag(qrow + 32, half);  // f = 32..63

    const __bf16* kgb = Kg  + (size_t)bh * GG * FF;
    const __bf16* klb = Kl  + (size_t)bh * SS * FF;
    const __bf16* vgb = Vgt + (size_t)bh * FF * GG;
    const __bf16* vlb = Vlt + (size_t)bh * FF * SS;
    const float*  mrow = mask + (size_t)b * SS;

    v8f o0 = {}, o1 = {}, o2 = {}, o3 = {};
    float mj[8], lj[8];
#pragma unroll
    for (int j = 0; j < 8; ++j) { mj[j] = -1e30f; lj[j] = 0.f; }

    const int total = GG + localLen;
    for (int kc = 0; kc < total; kc += 32) {
        // ---- scores for 32 keys (two 16-key tiles) ----
        v8f s0 = {}, s1 = {};
        const int key0 = kc + l15;
        const int key1 = key0 + 16;
        const __bf16* kr0 = (key0 < GG)
                                ? (kgb + (size_t)key0 * FF)
                                : (klb + (size_t)(lbase + key0 - GG) * FF);
        const __bf16* kr1 = (key1 < GG)
                                ? (kgb + (size_t)key1 * FF)
                                : (klb + (size_t)(lbase + key1 - GG) * FF);
        {
            v16bf b0 = load_bfrag(kr0 + half * 16);
            v16bf b1 = load_bfrag(kr0 + 32 + half * 16);
            s0 = wmma_bf16(qa0, b0, s0);
            s0 = wmma_bf16(qa1, b1, s0);
            v16bf b2 = load_bfrag(kr1 + half * 16);
            v16bf b3 = load_bfrag(kr1 + 32 + half * 16);
            s1 = wmma_bf16(qa0, b2, s1);
            s1 = wmma_bf16(qa1, b3, s1);
        }
        const float mv0 = (key0 >= GG) ? mrow[lbase + key0 - GG] : 0.f;
        const float mv1 = (key1 >= GG) ? mrow[lbase + key1 - GG] : 0.f;

        // ---- online softmax update, P -> LDS (bf16) ----
#pragma unroll
        for (int j = 0; j < 8; ++j) {
            float v0 = s0[j] + mv0;
            float v1 = s1[j] + mv1;
            float mx = fmaxf(v0, v1);
            mx = fmaxf(mx, __shfl_xor(mx, 1, 32));
            mx = fmaxf(mx, __shfl_xor(mx, 2, 32));
            mx = fmaxf(mx, __shfl_xor(mx, 4, 32));
            mx = fmaxf(mx, __shfl_xor(mx, 8, 32));
            float nm = fmaxf(mj[j], mx);
            float al = __expf(mj[j] - nm);
            float p0 = __expf(v0 - nm);
            float p1 = __expf(v1 - nm);
            float rs = p0 + p1;
            rs += __shfl_xor(rs, 1, 32);
            rs += __shfl_xor(rs, 2, 32);
            rs += __shfl_xor(rs, 4, 32);
            rs += __shfl_xor(rs, 8, 32);
            lj[j] = lj[j] * al + rs;
            mj[j] = nm;
            o0[j] *= al; o1[j] *= al; o2[j] *= al; o3[j] *= al;
            const int row = j + 8 * half;            // C-layout row
            lds_p[row * 32 + l15]      = f2bf(p0);   // keys kc..kc+15
            lds_p[row * 32 + 16 + l15] = f2bf(p1);   // keys kc+16..kc+31
        }
        asm volatile("s_wait_dscnt 0" ::: "memory");

        // ---- P (16x32) * V (32xF) ----
        const v16bf pa = load_afrag((const __bf16*)lds_p + (size_t)l15 * 32, half);
        const int kb = kc + half * 16;  // per-lane key base for B-frag
#pragma unroll
        for (int t = 0; t < 4; ++t) {
            const int fcol = t * 16 + l15;
            const __bf16* vr = (kb < GG)
                                   ? (vgb + (size_t)fcol * GG + kb)
                                   : (vlb + (size_t)fcol * SS + lbase + kb - GG);
            v16bf vb = load_bfrag(vr);
            if (t == 0) o0 = wmma_bf16(pa, vb, o0);
            else if (t == 1) o1 = wmma_bf16(pa, vb, o1);
            else if (t == 2) o2 = wmma_bf16(pa, vb, o2);
            else o3 = wmma_bf16(pa, vb, o3);
        }
    }

    // ---- normalize and write context (bf16, [B, Lq, D]) ----
#pragma unroll
    for (int j = 0; j < 8; ++j) {
        float inv = 1.0f / lj[j];
        int r = q0 + j + 8 * half;
        size_t rb = ((size_t)b * Lq + r) * DD + (size_t)h * FF;
        ctx[rb + 0  + l15] = f2bf(o0[j] * inv);
        ctx[rb + 16 + l15] = f2bf(o1[j] * inv);
        ctx[rb + 32 + l15] = f2bf(o2[j] * inv);
        ctx[rb + 48 + l15] = f2bf(o3[j] * inv);
    }
}

// ---------------------------------------------------------------------------
extern "C" void kernel_launch(void* const* d_in, const int* in_sizes, int n_in,
                              void* d_out, int out_size, void* d_ws, size_t ws_size,
                              hipStream_t stream) {
    const float* tok = (const float*)d_in[0]; // [B,S,D]
    const float* glb = (const float*)d_in[1]; // [B,G,D]
    const float* msk = (const float*)d_in[2]; // [B,S]
    const float* Wq  = (const float*)d_in[3];
    const float* Wk  = (const float*)d_in[4];
    const float* Wv  = (const float*)d_in[5];
    const float* Wo  = (const float*)d_in[6];
    float* out = (float*)d_out; // [B,S,D] local ++ [B,G,D] global

    const size_t nTok = (size_t)BB * SS * DD; // 8388608
    const size_t nGlb = (size_t)BB * GG * DD; // 262144
    const size_t nW   = (size_t)DD * DD;      // 1048576

    char* ws = (char*)d_ws;
    size_t off = 0;
    auto alloc = [&](size_t bytes) -> void* {
        void* p = ws + off;
        off = (off + bytes + 255) & ~(size_t)255;
        return p;
    };

    __bf16* tok_b = (__bf16*)alloc(nTok * 2);
    __bf16* glb_b = (__bf16*)alloc(nGlb * 2);
    __bf16* wq_b  = (__bf16*)alloc(nW * 2);
    __bf16* wk_b  = (__bf16*)alloc(nW * 2);
    __bf16* wv_b  = (__bf16*)alloc(nW * 2);
    __bf16* wo_b  = (__bf16*)alloc(nW * 2);
    __bf16* ql  = (__bf16*)alloc(nTok * 2); // [B,H,S,F]
    __bf16* kl  = (__bf16*)alloc(nTok * 2); // [B,H,S,F]
    __bf16* vlt = (__bf16*)alloc(nTok * 2); // [B,H,F,S]
    __bf16* qg  = (__bf16*)alloc(nGlb * 2); // [B,H,G,F]
    __bf16* kg  = (__bf16*)alloc(nGlb * 2); // [B,H,G,F]
    __bf16* vgt = (__bf16*)alloc(nGlb * 2); // [B,H,F,G]
    unsigned short* ctx_l = (unsigned short*)alloc(nTok * 2); // bf16 [B,S,D]
    unsigned short* ctx_g = (unsigned short*)alloc(nGlb * 2); // bf16 [B,G,D]

    // 1) convert everything to bf16
    k_cvt_bf16<<<(int)((nTok + 255) / 256), 256, 0, stream>>>(tok, (unsigned short*)tok_b, (int)nTok);
    k_cvt_bf16<<<(int)((nGlb + 255) / 256), 256, 0, stream>>>(glb, (unsigned short*)glb_b, (int)nGlb);
    k_cvt_bf16<<<(int)((nW + 255) / 256), 256, 0, stream>>>(Wq, (unsigned short*)wq_b, (int)nW);
    k_cvt_bf16<<<(int)((nW + 255) / 256), 256, 0, stream>>>(Wk, (unsigned short*)wk_b, (int)nW);
    k_cvt_bf16<<<(int)((nW + 255) / 256), 256, 0, stream>>>(Wv, (unsigned short*)wv_b, (int)nW);
    k_cvt_bf16<<<(int)((nW + 255) / 256), 256, 0, stream>>>(Wo, (unsigned short*)wo_b, (int)nW);

    // 2) QKV projections (WMMA GEMMs, A tile staged via TDM)
    dim3 blk(256);
    dim3 gt(DD / 512, (BB * SS) / 16);  // token rows: M = 8192
    dim3 gg(DD / 512, (BB * GG) / 16);  // global rows: M = 256
    k_gemm_bf16_wmma<<<gt, blk, 0, stream>>>(tok_b, wq_b, ql,  DD, DD, 12, 0, SCALING);
    k_gemm_bf16_wmma<<<gt, blk, 0, stream>>>(tok_b, wk_b, kl,  DD, DD, 12, 0, 1.0f);
    k_gemm_bf16_wmma<<<gt, blk, 0, stream>>>(tok_b, wv_b, vlt, DD, DD, 12, 1, 1.0f);
    k_gemm_bf16_wmma<<<gg, blk, 0, stream>>>(glb_b, wq_b, qg,  DD, DD, 7, 0, SCALING);
    k_gemm_bf16_wmma<<<gg, blk, 0, stream>>>(glb_b, wk_b, kg,  DD, DD, 7, 0, 1.0f);
    k_gemm_bf16_wmma<<<gg, blk, 0, stream>>>(glb_b, wv_b, vgt, DD, DD, 7, 1, 1.0f);

    // 3) attention (flash-style, WMMA)
    // global queries attend to [global ; all local] keys
    k_attn_wmma<<<BB * HH * (GG / 16), 32, 0, stream>>>(
        qg, kg, vgt, kl, vlt, msk, ctx_g, GG, SS, 0);
    // local blocked queries attend to [global ; same-block local] keys
    k_attn_wmma<<<BB * HH * (SS / 16), 32, 0, stream>>>(
        ql, kg, vgt, kl, vlt, msk, ctx_l, SS, KKB, 1);

    // 4) output projections (f32 straight into d_out)
    k_gemm_bf16_wmma<<<gt, blk, 0, stream>>>((const __bf16*)ctx_l, wo_b, out,
                                             DD, DD, 12, 2, 1.0f);
    k_gemm_bf16_wmma<<<gg, blk, 0, stream>>>((const __bf16*)ctx_g, wo_b, out + nTok,
                                             DD, DD, 7, 2, 1.0f);
}
